// ddbraingnn_68771016344263
// MI455X (gfx1250) — matrix operand order repaired and checked by reading, not compiled, not run on hardware
//
#include <hip/hip_runtime.h>

typedef __attribute__((ext_vector_type(16))) _Float16 v16h;
typedef __attribute__((ext_vector_type(4)))  _Float16 v4h;
typedef __attribute__((ext_vector_type(8)))  float    v8f;
typedef __attribute__((ext_vector_type(4), aligned(4))) float v4fu;  // 4B-aligned loads

#define KN0   111
#define KN1   56
#define KN2   28
#define KD2   128
#define KC    256
#define KDIN  8192
#define KNOUT 128
#define KCB   128

// ---------------- WMMA fragment layouts (CDNA5 wave32) ----------------
// A (16x32 f16): value (m,k) at lane = m + 16*((k>>3)&1), elem = 8*(k>>4) + (k&7)
//   -> 4 consecutive k (k%4==0) are contiguous LDS halves (ds_store_b64 ok)
// B (32x16 f16): value (k,n) at lane = 16*((k>>3)&1) + n, elem = 8*(k>>4) + (k&7)
// C/D (16x16 f32): vgpr r -> m = r + 8*(lane>=16); n = lane&15

struct GP {
  const float* A; long As; int lda; int Ar; int Ac;      // A matrix (batched)
  const float* Av; long Avs;                              // row-scale vec (SCALED)
  const float* Aw; long Aws;                              // col-scale vec (SCALED)
  float Amul;
  const float* B; long Bs; int ldb; int Br; int Bc;      // B matrix
  const float* C; long Cs; int ldc; float Cmul;          // accumulator init (HASC)
  float* D; long Ds; int ldd; int Dr; int Dc; int dcol0; int relu;
  int M, N, K, nbat;
};

__device__ inline int a_slot(int m, int k) {  // m in [0,32), two 512-half blocks
  return ((m >> 4) << 9) + (((m & 15) + (((k >> 3) & 1) << 4)) << 4) +
         ((k >> 4) << 3) + (k & 7);
}
__device__ inline int b_slot(int jj, int k, int n) {
  return (jj << 9) + (((((k >> 3) & 1) << 4) + n) << 4) + ((k >> 4) << 3) + (k & 7);
}

// Block = 128 threads / 4 waves; block tile = 32 rows x 64 cols; K-step 32.
// Each wave computes two stacked 16x16 tiles sharing one B fragment:
// 8 WMMAs per block k-step for 3072 staged elements.
template <bool SCALED, bool EYE, bool HASC>
__global__ void __launch_bounds__(128) gemm_t(GP p) {
  int mt = (p.M + 31) >> 5, ntb = (p.N + 63) >> 6;
  int tiles = mt * ntb;
  int bid = blockIdx.x;
  int b = bid / tiles, t = bid % tiles;
  int row0 = (t / ntb) << 5;
  int col0 = (t % ntb) << 6;
  const float* A = p.A + (long)b * p.As;
  const float* B = p.B + (long)b * p.Bs;
  const float* Av = SCALED ? p.Av + (long)b * p.Avs : nullptr;
  const float* Aw = SCALED ? p.Aw + (long)b * p.Aws : nullptr;

  __shared__ v16h sAv[64];   // 1024 halves: two 16x32 fragment blocks
  __shared__ v16h sBv[128];  // 4 tiles x 512 halves
  _Float16* sA = (_Float16*)sAv;
  _Float16* sB = (_Float16*)sBv;

  int tid = threadIdx.x;
  int wid = tid >> 5, lane = tid & 31, hf = lane >> 4, mr = lane & 15;
  int wcol0 = col0 + (wid << 4);

  v8f acc0, acc1;
#pragma unroll
  for (int r = 0; r < 8; ++r) {
    float c0 = 0.f, c1 = 0.f;
    if (HASC) {
      int m = row0 + r + (hf << 3), n = wcol0 + mr;
      if (m < p.M && n < p.N) c0 = p.Cmul * p.C[(long)b * p.Cs + m * p.ldc + n];
      if (m + 16 < p.M && n < p.N)
        c1 = p.Cmul * p.C[(long)b * p.Cs + (m + 16) * p.ldc + n];
    }
    acc0[r] = c0;
    acc1[r] = c1;
  }

  for (int k0 = 0; k0 < p.K; k0 += 32) {
    // ---- stage A (32x32) ----
    if (row0 + 32 <= p.Ar && k0 + 32 <= p.Ac) {  // fast: 2 quads per thread
#pragma unroll
      for (int it = 0; it < 2; ++it) {
        int idx = tid + (it << 7);
        int m = idx >> 3, k = (idx & 7) << 2;
        int gm = row0 + m, gk = k0 + k;
        v4fu a4 = *(const v4fu*)(A + gm * p.lda + gk);
        float rs = p.Amul;
        if (SCALED) rs *= Av[gm];
        v4h h4;
#pragma unroll
        for (int j = 0; j < 4; ++j) {
          float v = a4[j];
          if (EYE) v += (gm == gk + j) ? 1.f : 0.f;
          if (SCALED) v *= Aw[gk + j];
          h4[j] = (_Float16)(v * rs);
        }
        *(v4h*)&sA[a_slot(m, k)] = h4;
      }
    } else {
      for (int i = tid; i < 1024; i += 128) {
        int m = i >> 5, k = i & 31;
        int gm = row0 + m, gk = k0 + k;
        float v = 0.f;
        if (gm < p.Ar && gk < p.Ac) {
          v = A[gm * p.lda + gk];
          if (EYE) v += (gm == gk) ? 1.f : 0.f;
          if (SCALED) v *= Av[gm] * Aw[gk];
          v *= p.Amul;
        }
        sA[a_slot(m, k)] = (_Float16)v;
      }
    }
    // ---- stage B (32x64) ----
    if (k0 + 32 <= p.Br && col0 + 64 <= p.Bc) {  // fast: 4 quads per thread
#pragma unroll
      for (int it = 0; it < 4; ++it) {
        int idx = tid + (it << 7);
        int k = idx >> 4, c = (idx & 15) << 2;
        int gk = k0 + k, gn = col0 + c;
        v4fu b4 = *(const v4fu*)(B + gk * p.ldb + gn);
        int jj = c >> 4, n0 = c & 15;
#pragma unroll
        for (int j = 0; j < 4; ++j) sB[b_slot(jj, k, n0 + j)] = (_Float16)b4[j];
      }
    } else {
      for (int i = tid; i < 2048; i += 128) {
        int k = i >> 6, c = i & 63;
        int gk = k0 + k, gn = col0 + c;
        float v = (gk < p.Br && gn < p.Bc) ? B[gk * p.ldb + gn] : 0.f;
        sB[b_slot(c >> 4, k, c & 15)] = (_Float16)v;
      }
    }
    __syncthreads();
    v16h af0 = sAv[lane];
    v16h af1 = sAv[32 + lane];
    v16h bf = sBv[(wid << 5) + lane];
    acc0 = __builtin_amdgcn_wmma_f32_16x16x32_f16(false, af0, false, bf, (short)0, acc0,
                                                  false, false);
    acc1 = __builtin_amdgcn_wmma_f32_16x16x32_f16(false, af1, false, bf, (short)0, acc1,
                                                  false, false);
    __syncthreads();
  }

  float* D = p.D + (long)b * p.Ds;
#pragma unroll
  for (int r = 0; r < 8; ++r) {
    int m = row0 + r + (hf << 3), n = wcol0 + mr;
    if (m < p.Dr && n < p.Dc) {
      float v2 = acc0[r];
      if (p.relu) v2 = fmaxf(v2, 0.f);
      D[m * p.ldd + p.dcol0 + n] = v2;
    }
    if (m + 16 < p.Dr && n < p.Dc) {
      float v2 = acc1[r];
      if (p.relu) v2 = fmaxf(v2, 0.f);
      D[(m + 16) * p.ldd + p.dcol0 + n] = v2;
    }
  }
}

// Cheb combine: D = relu(A0*B0 + A1*B1 + A2*B2) at column offset dcol0.
struct CP {
  const float* A0; long A0s; int lda0;
  const float* A1; long A1s; int lda1;
  const float* A2; long A2s; int lda2;
  int Ar, Ac;
  const float* B0; const float* B1; const float* B2;
  int ldb, Br, Bc;
  float* D; long Ds; int ldd; int dcol0;
  int M, N, K, nbat;
};

__global__ void __launch_bounds__(128) cheb_kernel(CP p) {
  int mt = (p.M + 31) >> 5, ntb = (p.N + 63) >> 6;
  int tiles = mt * ntb;
  int bid = blockIdx.x;
  int b = bid / tiles, t = bid % tiles;
  int row0 = (t / ntb) << 5;
  int col0 = (t % ntb) << 6;

  __shared__ v16h sAv[64];
  __shared__ v16h sBv[128];
  _Float16* sA = (_Float16*)sAv;
  _Float16* sB = (_Float16*)sBv;

  int tid = threadIdx.x;
  int wid = tid >> 5, lane = tid & 31, hf = lane >> 4, mr = lane & 15;
  int wcol0 = col0 + (wid << 4);

  v8f acc0, acc1;
#pragma unroll
  for (int r = 0; r < 8; ++r) { acc0[r] = 0.f; acc1[r] = 0.f; }

  const float* As[3] = {p.A0 + (long)b * p.A0s, p.A1 + (long)b * p.A1s,
                        p.A2 + (long)b * p.A2s};
  const float* Bs[3] = {p.B0, p.B1, p.B2};
  int ldas[3] = {p.lda0, p.lda1, p.lda2};

  for (int ph = 0; ph < 3; ++ph) {
    const float* A = As[ph];
    const float* B = Bs[ph];
    int lda = ldas[ph];
    for (int k0 = 0; k0 < p.K; k0 += 32) {
      if (row0 + 32 <= p.Ar && k0 + 32 <= p.Ac) {
#pragma unroll
        for (int it = 0; it < 2; ++it) {
          int idx = tid + (it << 7);
          int m = idx >> 3, k = (idx & 7) << 2;
          v4fu a4 = *(const v4fu*)(A + (row0 + m) * lda + k0 + k);
          v4h h4;
#pragma unroll
          for (int j = 0; j < 4; ++j) h4[j] = (_Float16)a4[j];
          *(v4h*)&sA[a_slot(m, k)] = h4;
        }
      } else {
        for (int i = tid; i < 1024; i += 128) {
          int m = i >> 5, k = i & 31;
          int gm = row0 + m, gk = k0 + k;
          float v = (gm < p.Ar && gk < p.Ac) ? A[gm * lda + gk] : 0.f;
          sA[a_slot(m, k)] = (_Float16)v;
        }
      }
      if (k0 + 32 <= p.Br && col0 + 64 <= p.Bc) {
#pragma unroll
        for (int it = 0; it < 4; ++it) {
          int idx = tid + (it << 7);
          int k = idx >> 4, c = (idx & 15) << 2;
          v4fu b4 = *(const v4fu*)(B + (k0 + k) * p.ldb + col0 + c);
          int jj = c >> 4, n0 = c & 15;
#pragma unroll
          for (int j = 0; j < 4; ++j) sB[b_slot(jj, k, n0 + j)] = (_Float16)b4[j];
        }
      } else {
        for (int i = tid; i < 2048; i += 128) {
          int k = i >> 6, c = i & 63;
          int gk = k0 + k, gn = col0 + c;
          float v = (gk < p.Br && gn < p.Bc) ? B[gk * p.ldb + gn] : 0.f;
          sB[b_slot(c >> 4, k, c & 15)] = (_Float16)v;
        }
      }
      __syncthreads();
      v16h af0 = sAv[lane];
      v16h af1 = sAv[32 + lane];
      v16h bf = sBv[(wid << 5) + lane];
      acc0 = __builtin_amdgcn_wmma_f32_16x16x32_f16(false, af0, false, bf, (short)0,
                                                    acc0, false, false);
      acc1 = __builtin_amdgcn_wmma_f32_16x16x32_f16(false, af1, false, bf, (short)0,
                                                    acc1, false, false);
      __syncthreads();
    }
  }

  float* D = p.D + (long)b * p.Ds;
#pragma unroll
  for (int r = 0; r < 8; ++r) {
    int m = row0 + r + (hf << 3), n = wcol0 + mr;
    if (m < p.M && n < p.N) D[m * p.ldd + p.dcol0 + n] = fmaxf(acc0[r], 0.f);
    if (m + 16 < p.M && n < p.N)
      D[(m + 16) * p.ldd + p.dcol0 + n] = fmaxf(acc1[r], 0.f);
  }
}

// ---------------- degree prep for the input graphs ----------------
__global__ void __launch_bounds__(128) prep0_kernel(const float* adj, float* dinv,
                                                    float* dis, float* disa) {
  int b = blockIdx.x;
  const float* A = adj + (size_t)b * KN0 * KN0;
  for (int i = threadIdx.x; i < KN0; i += blockDim.x) {
    float d = 0.f;
    for (int j = 0; j < KN0; ++j) d += A[i * KN0 + j];
    dinv[b * KN0 + i] = d > 0.f ? 1.f / d : 0.f;
    dis[b * KN0 + i]  = d > 0.f ? rsqrtf(d) : 0.f;
    disa[b * KN0 + i] = rsqrtf(d + 1.f);
  }
}

__device__ inline float wave_sum(float s) {
#pragma unroll
  for (int off = 16; off >= 1; off >>= 1) s += __shfl_xor(s, off, 32);
  return s;
}

// ---------------- pool1: score, top-56, gather, readout x1, subgraph prep ----------------
__global__ void __launch_bounds__(128) pool1_kernel(const float* H1, const float* adj,
                                                    const float* DINV, float* Zc,
                                                    float* H1P, float* ADJ1, float* DINV1,
                                                    float* DIS1, float* DISA1) {
  int b = blockIdx.x;
  const float* H = H1 + (size_t)b * KN0 * KC;
  const float* A = adj + (size_t)b * KN0 * KN0;
  const float* dinv = DINV + (size_t)b * KN0;
  float* Z = Zc + (size_t)b * KDIN;
  float* Hp = H1P + (size_t)b * KN1 * KC;
  float* A1 = ADJ1 + (size_t)b * KN1 * KN1;
  __shared__ float s_score[KN0];
  __shared__ int s_idx[KN1];
  int wid = threadIdx.x >> 5, lane = threadIdx.x & 31;
  for (int i = wid; i < KN0; i += 4) {
    float acc[8];
#pragma unroll
    for (int t = 0; t < 8; ++t) acc[t] = 0.f;
    const float* ar = A + i * KN0;
    for (int j = 0; j < KN0; ++j) {
      float a = ar[j];
      const float* hr = H + j * KC;
#pragma unroll
      for (int t = 0; t < 8; ++t) acc[t] = fmaf(a, hr[lane + 32 * t], acc[t]);
    }
    float di = dinv[i];
    const float* hi = H + i * KC;
    float s = 0.f;
#pragma unroll
    for (int t = 0; t < 8; ++t) s += fabsf(hi[lane + 32 * t] - di * acc[t]);
    s = wave_sum(s);
    if (lane == 0) s_score[i] = s;
  }
  __syncthreads();
  if (threadIdx.x == 0) {  // stable top-k: strict '>' => lowest index wins ties
    unsigned long long u0 = 0, u1 = 0;
    for (int r = 0; r < KN1; ++r) {
      float bv = -1e30f;
      int bi = 0;
      for (int i = 0; i < KN0; ++i) {
        bool used = (i < 64) ? ((u0 >> i) & 1ull) : ((u1 >> (i - 64)) & 1ull);
        if (!used && s_score[i] > bv) { bv = s_score[i]; bi = i; }
      }
      if (bi < 64) u0 |= 1ull << bi; else u1 |= 1ull << (bi - 64);
      s_idx[r] = bi;
    }
  }
  __syncthreads();
  for (int e = threadIdx.x; e < KN1 * KC; e += blockDim.x) {
    int r = e >> 8, f = e & 255;
    Hp[e] = H[s_idx[r] * KC + f];
  }
  for (int e = threadIdx.x; e < KN1 * KN1; e += blockDim.x) {
    int r = e / KN1, c = e % KN1;
    A1[e] = A[s_idx[r] * KN0 + s_idx[c]];
  }
  __syncthreads();
  for (int f = threadIdx.x; f < KC; f += blockDim.x) {  // x1 readout [max | mean]
    float mx = Hp[f], sm = 0.f;
    for (int r = 0; r < KN1; ++r) {
      float v = Hp[r * KC + f];
      mx = fmaxf(mx, v);
      sm += v;
    }
    Z[KN2 * KC + f] = mx;
    Z[KN2 * KC + KC + f] = sm * (1.f / KN1);
  }
  for (int r = threadIdx.x; r < KN1; r += blockDim.x) {
    float d = 0.f;
    for (int c = 0; c < KN1; ++c) d += A1[r * KN1 + c];
    DINV1[b * KN1 + r] = d > 0.f ? 1.f / d : 0.f;
    DIS1[b * KN1 + r]  = d > 0.f ? rsqrtf(d) : 0.f;
    DISA1[b * KN1 + r] = rsqrtf(d + 1.f);
  }
}

// ---------------- pool2: score, top-28, gather into z flat, readout x2 ----------------
__global__ void __launch_bounds__(128) pool2_kernel(const float* H2, const float* ADJ1,
                                                    const float* DINV1, float* Zc) {
  int b = blockIdx.x;
  const float* H = H2 + (size_t)b * KN1 * KC;
  const float* A = ADJ1 + (size_t)b * KN1 * KN1;
  const float* dinv = DINV1 + (size_t)b * KN1;
  float* Z = Zc + (size_t)b * KDIN;
  __shared__ float s_score[KN1];
  __shared__ int s_idx[KN2];
  int wid = threadIdx.x >> 5, lane = threadIdx.x & 31;
  for (int i = wid; i < KN1; i += 4) {
    float acc[8];
#pragma unroll
    for (int t = 0; t < 8; ++t) acc[t] = 0.f;
    const float* ar = A + i * KN1;
    for (int j = 0; j < KN1; ++j) {
      float a = ar[j];
      const float* hr = H + j * KC;
#pragma unroll
      for (int t = 0; t < 8; ++t) acc[t] = fmaf(a, hr[lane + 32 * t], acc[t]);
    }
    float di = dinv[i];
    const float* hi = H + i * KC;
    float s = 0.f;
#pragma unroll
    for (int t = 0; t < 8; ++t) s += fabsf(hi[lane + 32 * t] - di * acc[t]);
    s = wave_sum(s);
    if (lane == 0) s_score[i] = s;
  }
  __syncthreads();
  if (threadIdx.x == 0) {
    unsigned long long used = 0;
    for (int r = 0; r < KN2; ++r) {
      float bv = -1e30f;
      int bi = 0;
      for (int i = 0; i < KN1; ++i)
        if (!((used >> i) & 1ull) && s_score[i] > bv) { bv = s_score[i]; bi = i; }
      used |= 1ull << bi;
      s_idx[r] = bi;
    }
  }
  __syncthreads();
  for (int e = threadIdx.x; e < KN2 * KC; e += blockDim.x) {  // flat h2p
    int r = e >> 8, f = e & 255;
    Z[e] = H[s_idx[r] * KC + f];
  }
  __syncthreads();
  for (int f = threadIdx.x; f < KC; f += blockDim.x) {  // x2 readout
    float mx = Z[f], sm = 0.f;
    for (int r = 0; r < KN2; ++r) {
      float v = Z[r * KC + f];
      mx = fmaxf(mx, v);
      sm += v;
    }
    Z[KN2 * KC + 2 * KC + f] = mx;
    Z[KN2 * KC + 3 * KC + f] = sm * (1.f / KN2);
  }
}

// ---------------- softmax over features rows ----------------
__global__ void __launch_bounds__(128) softmax_kernel(const float* feat, float* xlo, int B) {
  int row = blockIdx.x * 4 + (threadIdx.x >> 5);
  if (row >= B) return;
  int lane = threadIdx.x & 31;
  const float* f = feat + (size_t)row * KNOUT;
  float v[4];
  float mx = -1e30f;
#pragma unroll
  for (int t = 0; t < 4; ++t) {
    v[t] = f[lane + 32 * t];
    mx = fmaxf(mx, v[t]);
  }
#pragma unroll
  for (int off = 16; off >= 1; off >>= 1) mx = fmaxf(mx, __shfl_xor(mx, off, 32));
  float s = 0.f;
#pragma unroll
  for (int t = 0; t < 4; ++t) {
    v[t] = __expf(v[t] - mx);
    s += v[t];
  }
  s = wave_sum(s);
  float inv = 1.f / s;
#pragma unroll
  for (int t = 0; t < 4; ++t) xlo[(size_t)row * KNOUT + lane + 32 * t] = v[t] * inv;
}

// ---------------- host driver ----------------
static inline int gemm_blocks(int M, int N, int nbat) {
  return nbat * ((M + 31) / 32) * ((N + 63) / 64);
}

extern "C" void kernel_launch(void* const* d_in, const int* in_sizes, int n_in,
                              void* d_out, int out_size, void* d_ws, size_t ws_size,
                              hipStream_t stream) {
  const float* x      = (const float*)d_in[0];
  const float* adj    = (const float*)d_in[1];
  const float* w_gcn1 = (const float*)d_in[2];
  const float* w_sp1  = (const float*)d_in[3];
  const float* w_gcn2 = (const float*)d_in[4];
  const float* w_sp2  = (const float*)d_in[5];
  const float* lin1_w = (const float*)d_in[6];
  const float* lin1_b = (const float*)d_in[7];
  const float* lin2_w = (const float*)d_in[8];
  const float* lin2_b = (const float*)d_in[9];
  float* out = (float*)d_out;
  int Btot = out_size / (2 * KNOUT);
  float* outF = out + (size_t)Btot * KNOUT;  // features region (2nd tuple output)

  // workspace carve (per chunk of KCB graphs)
  float* w = (float*)d_ws;
  size_t off = 0;
  auto alloc = [&](size_t n) { float* p = w + off; off += n; return p; };
  float* DINV  = alloc((size_t)KCB * KN0);
  float* DIS   = alloc((size_t)KCB * KN0);
  float* DISA  = alloc((size_t)KCB * KN0);
  float* T1    = alloc((size_t)KCB * KN1 * KC);   // holds 111x111 (ld 111) or 56x256
  float* T2    = alloc((size_t)KCB * KN1 * KC);
  float* XW    = alloc((size_t)KCB * KN0 * KD2);
  float* H1    = alloc((size_t)KCB * KN0 * KC);
  float* H1P   = alloc((size_t)KCB * KN1 * KC);
  float* ADJ1  = alloc((size_t)KCB * KN1 * KN1);
  float* DINV1 = alloc((size_t)KCB * KN1);
  float* DIS1  = alloc((size_t)KCB * KN1);
  float* DISA1 = alloc((size_t)KCB * KN1);
  float* H2    = alloc((size_t)KCB * KN1 * KC);
  float* Zc    = alloc((size_t)KCB * KDIN);
  float* F1    = alloc((size_t)KCB * KC);
  (void)ws_size; (void)in_sizes; (void)n_in;

  const long XAs = (long)KN0 * KN0;      // 12321
  const long Ts  = (long)KN1 * KC;       // 14336
  const long H1s = (long)KN0 * KC;       // 28416
  const long XWs = (long)KN0 * KD2;      // 14208

  for (int bb0 = 0; bb0 < Btot; bb0 += KCB) {
    int cb = (Btot - bb0 < KCB) ? (Btot - bb0) : KCB;
    const float* xb = x + (size_t)bb0 * XAs;
    const float* ab = adj + (size_t)bb0 * XAs;

    prep0_kernel<<<cb, 128, 0, stream>>>(ab, DINV, DIS, DISA);

    // t1 = lh @ x,   lh = -dis (.) adj (.) dis
    { GP p{}; p.A = ab; p.As = XAs; p.lda = KN0; p.Ar = KN0; p.Ac = KN0;
      p.Av = DIS; p.Avs = KN0; p.Aw = DIS; p.Aws = KN0; p.Amul = -1.f;
      p.B = xb; p.Bs = XAs; p.ldb = KN0; p.Br = KN0; p.Bc = KN0;
      p.D = T1; p.Ds = Ts; p.ldd = KN0; p.Dr = KN0; p.Dc = KN0;
      p.dcol0 = 0; p.relu = 0; p.M = KN0; p.N = KN0; p.K = KN0; p.nbat = cb;
      gemm_t<true, false, false><<<gemm_blocks(p.M, p.N, cb), 128, 0, stream>>>(p); }

    // t2 = 2*lh @ t1 - x   (fold factor 2 into A, C-init = -x)
    { GP p{}; p.A = ab; p.As = XAs; p.lda = KN0; p.Ar = KN0; p.Ac = KN0;
      p.Av = DIS; p.Avs = KN0; p.Aw = DIS; p.Aws = KN0; p.Amul = -2.f;
      p.B = T1; p.Bs = Ts; p.ldb = KN0; p.Br = KN0; p.Bc = KN0;
      p.C = xb; p.Cs = XAs; p.ldc = KN0; p.Cmul = -1.f;
      p.D = T2; p.Ds = Ts; p.ldd = KN0; p.Dr = KN0; p.Dc = KN0;
      p.dcol0 = 0; p.relu = 0; p.M = KN0; p.N = KN0; p.K = KN0; p.nbat = cb;
      gemm_t<true, false, true><<<gemm_blocks(p.M, p.N, cb), 128, 0, stream>>>(p); }

    // xw = x @ w_gcn1
    { GP p{}; p.A = xb; p.As = XAs; p.lda = KN0; p.Ar = KN0; p.Ac = KN0; p.Amul = 1.f;
      p.B = w_gcn1; p.Bs = 0; p.ldb = KD2; p.Br = KN0; p.Bc = KD2;
      p.D = XW; p.Ds = XWs; p.ldd = KD2; p.Dr = KN0; p.Dc = KD2;
      p.dcol0 = 0; p.relu = 0; p.M = KN0; p.N = KD2; p.K = KN0; p.nbat = cb;
      gemm_t<false, false, false><<<gemm_blocks(p.M, p.N, cb), 128, 0, stream>>>(p); }

    // h1[:, :128] = relu(an @ xw),  an = disA (.) (adj + I) (.) disA
    { GP p{}; p.A = ab; p.As = XAs; p.lda = KN0; p.Ar = KN0; p.Ac = KN0;
      p.Av = DISA; p.Avs = KN0; p.Aw = DISA; p.Aws = KN0; p.Amul = 1.f;
      p.B = XW; p.Bs = XWs; p.ldb = KD2; p.Br = KN0; p.Bc = KD2;
      p.D = H1; p.Ds = H1s; p.ldd = KC; p.Dr = KN0; p.Dc = KD2;
      p.dcol0 = 0; p.relu = 1; p.M = KN0; p.N = KD2; p.K = KN0; p.nbat = cb;
      gemm_t<true, true, false><<<gemm_blocks(p.M, p.N, cb), 128, 0, stream>>>(p); }

    // h1[:, 128:] = relu(x@wk0 + t1@wk1 + t2@wk2)
    { CP p{}; p.A0 = xb; p.A0s = XAs; p.lda0 = KN0;
      p.A1 = T1; p.A1s = Ts; p.lda1 = KN0;
      p.A2 = T2; p.A2s = Ts; p.lda2 = KN0;
      p.Ar = KN0; p.Ac = KN0;
      p.B0 = w_sp1; p.B1 = w_sp1 + (size_t)KN0 * KD2; p.B2 = w_sp1 + (size_t)2 * KN0 * KD2;
      p.ldb = KD2; p.Br = KN0; p.Bc = KD2;
      p.D = H1; p.Ds = H1s; p.ldd = KC; p.dcol0 = KD2;
      p.M = KN0; p.N = KD2; p.K = KN0; p.nbat = cb;
      cheb_kernel<<<gemm_blocks(p.M, p.N, cb), 128, 0, stream>>>(p); }

    pool1_kernel<<<cb, 128, 0, stream>>>(H1, ab, DINV, Zc, H1P, ADJ1, DINV1, DIS1, DISA1);

    // t1b = lh1 @ h1p
    { GP p{}; p.A = ADJ1; p.As = (long)KN1 * KN1; p.lda = KN1; p.Ar = KN1; p.Ac = KN1;
      p.Av = DIS1; p.Avs = KN1; p.Aw = DIS1; p.Aws = KN1; p.Amul = -1.f;
      p.B = H1P; p.Bs = Ts; p.ldb = KC; p.Br = KN1; p.Bc = KC;
      p.D = T1; p.Ds = Ts; p.ldd = KC; p.Dr = KN1; p.Dc = KC;
      p.dcol0 = 0; p.relu = 0; p.M = KN1; p.N = KC; p.K = KN1; p.nbat = cb;
      gemm_t<true, false, false><<<gemm_blocks(p.M, p.N, cb), 128, 0, stream>>>(p); }

    // t2b = 2*lh1 @ t1b - h1p
    { GP p{}; p.A = ADJ1; p.As = (long)KN1 * KN1; p.lda = KN1; p.Ar = KN1; p.Ac = KN1;
      p.Av = DIS1; p.Avs = KN1; p.Aw = DIS1; p.Aws = KN1; p.Amul = -2.f;
      p.B = T1; p.Bs = Ts; p.ldb = KC; p.Br = KN1; p.Bc = KC;
      p.C = H1P; p.Cs = Ts; p.ldc = KC; p.Cmul = -1.f;
      p.D = T2; p.Ds = Ts; p.ldd = KC; p.Dr = KN1; p.Dc = KC;
      p.dcol0 = 0; p.relu = 0; p.M = KN1; p.N = KC; p.K = KN1; p.nbat = cb;
      gemm_t<true, false, true><<<gemm_blocks(p.M, p.N, cb), 128, 0, stream>>>(p); }

    // xw2 = h1p @ w_gcn2
    { GP p{}; p.A = H1P; p.As = Ts; p.lda = KC; p.Ar = KN1; p.Ac = KC; p.Amul = 1.f;
      p.B = w_gcn2; p.Bs = 0; p.ldb = KD2; p.Br = KC; p.Bc = KD2;
      p.D = XW; p.Ds = XWs; p.ldd = KD2; p.Dr = KN1; p.Dc = KD2;
      p.dcol0 = 0; p.relu = 0; p.M = KN1; p.N = KD2; p.K = KC; p.nbat = cb;
      gemm_t<false, false, false><<<gemm_blocks(p.M, p.N, cb), 128, 0, stream>>>(p); }

    // h2[:, :128] = relu(an1 @ xw2)
    { GP p{}; p.A = ADJ1; p.As = (long)KN1 * KN1; p.lda = KN1; p.Ar = KN1; p.Ac = KN1;
      p.Av = DISA1; p.Avs = KN1; p.Aw = DISA1; p.Aws = KN1; p.Amul = 1.f;
      p.B = XW; p.Bs = XWs; p.ldb = KD2; p.Br = KN1; p.Bc = KD2;
      p.D = H2; p.Ds = Ts; p.ldd = KC; p.Dr = KN1; p.Dc = KD2;
      p.dcol0 = 0; p.relu = 1; p.M = KN1; p.N = KD2; p.K = KN1; p.nbat = cb;
      gemm_t<true, true, false><<<gemm_blocks(p.M, p.N, cb), 128, 0, stream>>>(p); }

    // h2[:, 128:] = relu(h1p@wk0 + t1b@wk1 + t2b@wk2)
    { CP p{}; p.A0 = H1P; p.A0s = Ts; p.lda0 = KC;
      p.A1 = T1; p.A1s = Ts; p.lda1 = KC;
      p.A2 = T2; p.A2s = Ts; p.lda2 = KC;
      p.Ar = KN1; p.Ac = KC;
      p.B0 = w_sp2; p.B1 = w_sp2 + (size_t)KC * KD2; p.B2 = w_sp2 + (size_t)2 * KC * KD2;
      p.ldb = KD2; p.Br = KC; p.Bc = KD2;
      p.D = H2; p.Ds = Ts; p.ldd = KC; p.dcol0 = KD2;
      p.M = KN1; p.N = KD2; p.K = KC; p.nbat = cb;
      cheb_kernel<<<gemm_blocks(p.M, p.N, cb), 128, 0, stream>>>(p); }

    pool2_kernel<<<cb, 128, 0, stream>>>(H2, ADJ1, DINV1, Zc);

    // f = relu(z @ lin1_w + b1): batch rows are M
    { GP p{}; p.A = Zc; p.As = 0; p.lda = KDIN; p.Ar = cb; p.Ac = KDIN; p.Amul = 1.f;
      p.B = lin1_w; p.Bs = 0; p.ldb = KC; p.Br = KDIN; p.Bc = KC;
      p.C = lin1_b; p.Cs = 0; p.ldc = 0; p.Cmul = 1.f;
      p.D = F1; p.Ds = 0; p.ldd = KC; p.Dr = cb; p.Dc = KC;
      p.dcol0 = 0; p.relu = 1; p.M = cb; p.N = KC; p.K = KDIN; p.nbat = 1;
      gemm_t<false, false, true><<<gemm_blocks(p.M, p.N, 1), 128, 0, stream>>>(p); }

    // features = relu(f @ lin2_w + b2) -> second output region
    { GP p{}; p.A = F1; p.As = 0; p.lda = KC; p.Ar = cb; p.Ac = KC; p.Amul = 1.f;
      p.B = lin2_w; p.Bs = 0; p.ldb = KNOUT; p.Br = KC; p.Bc = KNOUT;
      p.C = lin2_b; p.Cs = 0; p.ldc = 0; p.Cmul = 1.f;
      p.D = outF + (size_t)bb0 * KNOUT; p.Ds = 0; p.ldd = KNOUT; p.Dr = cb; p.Dc = KNOUT;
      p.dcol0 = 0; p.relu = 1; p.M = cb; p.N = KNOUT; p.K = KC; p.nbat = 1;
      gemm_t<false, false, true><<<gemm_blocks(p.M, p.N, 1), 128, 0, stream>>>(p); }
  }

  softmax_kernel<<<(Btot + 3) / 4, 128, 0, stream>>>(outF, out, Btot);
}
